// MultiSizeProjHead_68564857914039
// MI455X (gfx1250) — matrix operand order repaired and checked by reading, not compiled, not run on hardware
//
#include <hip/hip_runtime.h>
#include <cstdint>
#include <cstddef>

// ---------------------------------------------------------------------------
// MultiSizeProjHead for MI455X (gfx1250, wave32, WMMA)
//
//   xs   = x  @ W_in^T     + b_in        (32768 x 4096, K=1024)
//   y672 = xs @ W_shared^T + b_shared    (32768 x 672,  K=4096)
//   r672 = x  @ W_res^T    + b_res       (32768 x 672,  K=1024)
//   out[m, t<T] = silu(interp(y672[m],T)[t]) + interp(r672[m],T)[t]
//
// All GEMM operands are pre-split once into bf16 hi/lo planes, so the GEMM
// hot loop is pure global_load_b128 + v_wmma_f32_16x16x32_bf16 (3 wmma per
// tile: hi*hi + lo*hi + hi*lo ~ fp32 accuracy at bf16 matrix rate).
// ---------------------------------------------------------------------------

typedef __attribute__((ext_vector_type(16))) __bf16 v16bf;
typedef __attribute__((ext_vector_type(8)))  __bf16 v8bf;
typedef __attribute__((ext_vector_type(4)))  __bf16 v4bf;
typedef __attribute__((ext_vector_type(8)))  float  v8f;

#define EMBED   1024
#define INTER   4096
#define OUT_B   672
#define OUT_MAX 1344
#define M_TOT   (256 * 128)

// ---------------- elementwise fp32 -> (bf16 hi, bf16 lo) split ----------------
__global__ __launch_bounds__(256)
void msph_cvt_split(const float* __restrict__ src,
                    __bf16* __restrict__ hi, __bf16* __restrict__ lo,
                    long n4) {            // n4 = n/4, n multiple of 4
  const long i = (long)blockIdx.x * blockDim.x + threadIdx.x;
  if (i >= n4) return;
  const float4 v = *(const float4*)(src + 4 * i);
  float f[4] = {v.x, v.y, v.z, v.w};
  __bf16 h[4], l[4];
#pragma unroll
  for (int q = 0; q < 4; ++q) {
    h[q] = (__bf16)f[q];
    l[q] = (__bf16)(f[q] - (float)h[q]);
  }
  *(v4bf*)(hi + 4 * i) = (v4bf){h[0], h[1], h[2], h[3]};
  *(v4bf*)(lo + 4 * i) = (v4bf){l[0], l[1], l[2], l[3]};
}

// ---------------- GEMM: C[M,N] = A[M,K] * B[N,K]^T + bias[N] ----------------
// 256 threads = 8 waves; block tile 128x128; wave tile 32(M) x 64(N)
// = 2x4 grid of 16x16 wmma tiles; K-step 32, software-pipelined by 2.

struct Frags {
  v16bf ah[2], al[2];
  v16bf bh[4], bl[4];
};

// A-frag per lane (g = lane>>4, m = lane&15): halves 0..7 -> K = k0+8g+[0..8),
// halves 8..15 -> K = k0+16+8g+[0..8)
__device__ __forceinline__ v16bf load_a(const __bf16* __restrict__ P, int k0) {
  const __bf16* p = P + k0;
  v8bf a = *(const v8bf*)p;
  v8bf b = *(const v8bf*)(p + 16);
  return __builtin_shufflevector(a, b, 0, 1, 2, 3, 4, 5, 6, 7,
                                 8, 9, 10, 11, 12, 13, 14, 15);
}

// B-frag per lane (g, n = lane&15): halves 0..7 -> K = k0+8g+[0..8),
// halves 8..15 -> K = k0+16+8g+[0..8)
__device__ __forceinline__ v16bf load_b(const __bf16* __restrict__ P, int k0) {
  const __bf16* p = P + k0;
  v8bf a = *(const v8bf*)p;
  v8bf b = *(const v8bf*)(p + 16);
  return __builtin_shufflevector(a, b, 0, 1, 2, 3, 4, 5, 6, 7,
                                 8, 9, 10, 11, 12, 13, 14, 15);
}

template <bool SPLIT_OUT>
__global__ __launch_bounds__(256)
void msph_gemm(const __bf16* __restrict__ Ah, const __bf16* __restrict__ Al,
               const __bf16* __restrict__ Bh, const __bf16* __restrict__ Bl,
               const float* __restrict__ bias,
               float*  __restrict__ Cf,               // SPLIT_OUT == false
               __bf16* __restrict__ Ch, __bf16* __restrict__ Cl,  // == true
               int M, int N, int K) {
  const int lane = threadIdx.x & 31;
  const int wave = threadIdx.x >> 5;
  const int wm   = wave & 3;
  const int wn   = wave >> 2;
  const int g    = lane >> 4;
  const int ln   = lane & 15;

  const int bm = blockIdx.y * 128 + wm * 32;
  const int bn = blockIdx.x * 128 + wn * 64;

  // Per-lane base pointers (all loads in-bounds: out-of-range columns are
  // clamped to N-1; their accumulators are computed but never stored).
  const __bf16* pAh[2];
  const __bf16* pAl[2];
#pragma unroll
  for (int i = 0; i < 2; ++i) {
    const size_t rowoff = (size_t)(bm + i * 16 + ln) * (size_t)K + 8 * g;
    pAh[i] = Ah + rowoff;
    pAl[i] = Al + rowoff;
  }
  int  cols[4];
  bool cok[4];
  const __bf16* pBh[4];
  const __bf16* pBl[4];
#pragma unroll
  for (int j = 0; j < 4; ++j) {
    cols[j] = bn + j * 16 + ln;
    cok[j]  = cols[j] < N;
    const int cc = cok[j] ? cols[j] : (N - 1);
    const size_t coloff = (size_t)cc * (size_t)K + 8 * g;
    pBh[j] = Bh + coloff;
    pBl[j] = Bl + coloff;
  }

  v8f acc[2][4];
#pragma unroll
  for (int i = 0; i < 2; ++i)
#pragma unroll
    for (int j = 0; j < 4; ++j)
      acc[i][j] = (v8f){0.f, 0.f, 0.f, 0.f, 0.f, 0.f, 0.f, 0.f};

  Frags F0, F1;

  auto loadF = [&](Frags& F, int k0) {
#pragma unroll
    for (int i = 0; i < 2; ++i) {
      F.ah[i] = load_a(pAh[i], k0);
      F.al[i] = load_a(pAl[i], k0);
    }
#pragma unroll
    for (int j = 0; j < 4; ++j) {
      F.bh[j] = load_b(pBh[j], k0);
      F.bl[j] = load_b(pBl[j], k0);
    }
  };

  auto computeF = [&](const Frags& F) {
#pragma unroll
    for (int i = 0; i < 2; ++i) {
#pragma unroll
      for (int j = 0; j < 4; ++j) {
        acc[i][j] = __builtin_amdgcn_wmma_f32_16x16x32_bf16(
            false, F.ah[i], false, F.bh[j], (short)0, acc[i][j], false, false);
        acc[i][j] = __builtin_amdgcn_wmma_f32_16x16x32_bf16(
            false, F.al[i], false, F.bh[j], (short)0, acc[i][j], false, false);
        acc[i][j] = __builtin_amdgcn_wmma_f32_16x16x32_bf16(
            false, F.ah[i], false, F.bl[j], (short)0, acc[i][j], false, false);
      }
    }
  };

  // 2-stage software pipeline; K % 64 == 0 and K >= 128 (1024 or 4096).
  loadF(F0, 0);
  int k0 = 0;
  for (; k0 < K - 64; k0 += 64) {
    loadF(F1, k0 + 32);
    computeF(F0);
    loadF(F0, k0 + 64);
    computeF(F1);
  }
  loadF(F1, k0 + 32);
  computeF(F0);
  computeF(F1);

  // store: D layout -> VGPR r, lane (g, n): element (M = r + 8g, N = n)
#pragma unroll
  for (int j = 0; j < 4; ++j) {
    if (!cok[j]) continue;
    const int col = cols[j];
    const float bv = bias[col];
#pragma unroll
    for (int i = 0; i < 2; ++i) {
      const int rowbase = bm + i * 16 + 8 * g;
#pragma unroll
      for (int r = 0; r < 8; ++r) {
        const size_t idx = (size_t)(rowbase + r) * (size_t)N + col;
        const float v = acc[i][j][r] + bv;
        if (SPLIT_OUT) {
          const __bf16 h = (__bf16)v;
          Ch[idx] = h;
          Cl[idx] = (__bf16)(v - (float)h);
        } else {
          Cf[idx] = v;
        }
      }
    }
  }
}

// ---------------- epilogue: interp + silu + residual + zero-pad ----------------
__global__ __launch_bounds__(256)
void msph_epilogue(const float* __restrict__ y672,
                   const float* __restrict__ r672,
                   const int* __restrict__ patch_sizes,
                   float* __restrict__ out,   // chunk base
                   int rows, int m0) {
  const int idx = blockIdx.x * blockDim.x + threadIdx.x;
  const int total = rows * OUT_MAX;
  if (idx >= total) return;

  const int m = idx / OUT_MAX;
  const int t = idx - m * OUT_MAX;
  const int n_sample = (m0 + m) >> 7;          // / 128 tokens per sample
  const int T = patch_sizes[n_sample] * 21;

  float o = 0.f;
  if (t < T) {
    const float scale = (float)OUT_B / (float)T;
    float src = ((float)t + 0.5f) * scale - 0.5f;
    src = fmaxf(src, 0.f);
    int i0 = (int)src;
    if (i0 > OUT_B - 1) i0 = OUT_B - 1;
    const int i1 = (i0 + 1 < OUT_B) ? i0 + 1 : OUT_B - 1;
    const float lam = src - (float)i0;

    const float* yr = y672 + (size_t)m * OUT_B;
    const float* rr = r672 + (size_t)m * OUT_B;
    const float yv = yr[i0] * (1.f - lam) + yr[i1] * lam;
    const float rv = rr[i0] * (1.f - lam) + rr[i1] * lam;
    o = yv / (1.f + expf(-yv)) + rv;
  }
  out[idx] = o;
}

// ---------------------------------------------------------------------------
extern "C" void kernel_launch(void* const* d_in, const int* in_sizes, int n_in,
                              void* d_out, int out_size, void* d_ws, size_t ws_size,
                              hipStream_t stream) {
  (void)in_sizes; (void)n_in; (void)out_size;

  const float* x     = (const float*)d_in[0];
  const int*   psz   = (const int*)d_in[1];
  const float* W_in  = (const float*)d_in[2];
  const float* b_in  = (const float*)d_in[3];
  const float* W_sh  = (const float*)d_in[4];
  const float* b_sh  = (const float*)d_in[5];
  const float* W_res = (const float*)d_in[6];
  const float* b_res = (const float*)d_in[7];
  float* out = (float*)d_out;

  // ---- workspace layout -------------------------------------------------
  char* wsb = (char*)d_ws;
  size_t off = 0;
  auto carve = [&](size_t bytes) {
    char* p = wsb + off;
    off += (bytes + 255) & ~(size_t)255;
    return p;
  };
  const size_t nx   = (size_t)M_TOT * EMBED;   // 33.5M
  const size_t nwi  = (size_t)INTER * EMBED;
  const size_t nws  = (size_t)OUT_B * INTER;
  const size_t nwr  = (size_t)OUT_B * EMBED;

  __bf16* x_h   = (__bf16*)carve(nx  * 2);
  __bf16* x_l   = (__bf16*)carve(nx  * 2);
  __bf16* wi_h  = (__bf16*)carve(nwi * 2);
  __bf16* wi_l  = (__bf16*)carve(nwi * 2);
  __bf16* wsh_h = (__bf16*)carve(nws * 2);
  __bf16* wsh_l = (__bf16*)carve(nws * 2);
  __bf16* wr_h  = (__bf16*)carve(nwr * 2);
  __bf16* wr_l  = (__bf16*)carve(nwr * 2);

  // chunked region: xs hi/lo (bf16) + y/r (fp32)
  const size_t perRow = (size_t)INTER * 2 * 2 + (size_t)OUT_B * 4 * 2;  // 21760
  long cm = 0;
  if (ws_size > off + 8192) cm = (long)((ws_size - off - 8192) / perRow);
  cm = (cm / 128) * 128;
  if (cm < 128) cm = 128;
  if (cm > M_TOT) cm = M_TOT;
  const int CM = (int)cm;

  __bf16* xs_h = (__bf16*)carve((size_t)CM * INTER * 2);
  __bf16* xs_l = (__bf16*)carve((size_t)CM * INTER * 2);
  float*  yb   = (float*)carve((size_t)CM * OUT_B * 4);
  float*  rb   = (float*)carve((size_t)CM * OUT_B * 4);

  dim3 blk(256);

  // ---- one-time splits (pure bandwidth) ---------------------------------
  auto split = [&](const float* s, __bf16* h, __bf16* l, size_t n) {
    const long n4 = (long)(n / 4);
    msph_cvt_split<<<(unsigned)((n4 + 255) / 256), blk, 0, stream>>>(s, h, l, n4);
  };
  split(x,     x_h,   x_l,   nx);
  split(W_in,  wi_h,  wi_l,  nwi);
  split(W_sh,  wsh_h, wsh_l, nws);
  split(W_res, wr_h,  wr_l,  nwr);

  // ---- chunked pipeline --------------------------------------------------
  for (int m0 = 0; m0 < M_TOT; m0 += CM) {
    const int rows = (M_TOT - m0 < CM) ? (M_TOT - m0) : CM;  // multiple of 128

    // xs = x_chunk @ W_in^T + b_in      (split bf16 output)
    dim3 g1(INTER / 128, rows / 128);
    msph_gemm<true><<<g1, blk, 0, stream>>>(
        x_h + (size_t)m0 * EMBED, x_l + (size_t)m0 * EMBED,
        wi_h, wi_l, b_in, nullptr, xs_h, xs_l, rows, INTER, EMBED);

    // y672 = xs @ W_shared^T + b_shared
    dim3 g2((OUT_B + 127) / 128, rows / 128);
    msph_gemm<false><<<g2, blk, 0, stream>>>(
        xs_h, xs_l, wsh_h, wsh_l, b_sh, yb, nullptr, nullptr,
        rows, OUT_B, INTER);

    // r672 = x_chunk @ W_res^T + b_res
    msph_gemm<false><<<g2, blk, 0, stream>>>(
        x_h + (size_t)m0 * EMBED, x_l + (size_t)m0 * EMBED,
        wr_h, wr_l, b_res, rb, nullptr, nullptr,
        rows, OUT_B, EMBED);

    // interp + silu + residual + zero-pad
    const int total = rows * OUT_MAX;
    msph_epilogue<<<(unsigned)((total + 255) / 256), blk, 0, stream>>>(
        yb, rb, psz, out + (size_t)m0 * OUT_MAX, rows, m0);
  }
}
